// GATImpactPredictor_34600256537003
// MI455X (gfx1250) — matrix operand order repaired and checked by reading, not compiled, not run on hardware
//
#include <hip/hip_runtime.h>
#include <hip/hip_bf16.h>
#include <math.h>

// ---------------- types ----------------
typedef _Float16 f16;
typedef __attribute__((ext_vector_type(16))) _Float16 v16h;
typedef __attribute__((ext_vector_type(8)))  _Float16 v8h;
typedef __attribute__((ext_vector_type(8)))  float    v8f;

// exact parameter types for __builtin_amdgcn_global_load_async_to_lds_b128
// (from hipcc diagnostic: "int __vector(4) __device__ *")
typedef int v4i __attribute__((vector_size(16)));
typedef __attribute__((address_space(1))) v4i* gv4i_p;   // global  <4 x i32>*
typedef __attribute__((address_space(3))) v4i* lv4i_p;   // LDS     <4 x i32>*

#define HEADS1 8
#define C1_    64
#define C2_    32
#define DIN    256          // x feature dim (K of GEMM1 minus the mod column)
#define F1     512          // H*C1
#define KT     32           // WMMA k-tile
#define LDAH   40           // padded LDS row in halves (80B -> destriped banks, 16B aligned chunks)

// CDNA5 async global->LDS staging, guarded so an availability miss
// falls back to the known-good synchronous path.
#ifndef USE_ASYNC_COPY
#  if defined(__has_builtin)
#    if __has_builtin(__builtin_amdgcn_global_load_async_to_lds_b128)
#      define USE_ASYNC_COPY 1
#    else
#      define USE_ASYNC_COPY 0
#    endif
#  else
#    define USE_ASYNC_COPY 0
#  endif
#endif

__device__ __forceinline__ void async_wait_all() {
#if USE_ASYNC_COPY
#  if defined(__has_builtin) && __has_builtin(__builtin_amdgcn_s_wait_asynccnt)
  __builtin_amdgcn_s_wait_asynccnt(0);
#  else
  asm volatile("s_wait_asynccnt 0x0" ::: "memory");
#  endif
#endif
}

__device__ __forceinline__ void copy16_g2l(const f16* src, f16* dstLds) {
#if USE_ASYNC_COPY
  __builtin_amdgcn_global_load_async_to_lds_b128(
      (gv4i_p)(uintptr_t)src, (lv4i_p)(unsigned)(uintptr_t)dstLds, 0, 0);
#else
  *(v8h*)dstLds = *(const v8h*)src;
#endif
}

__device__ __forceinline__ float leaky02(float x) { return x > 0.f ? x : 0.2f * x; }

__device__ __forceinline__ void atomicMaxF(float* addr, float v) {
  if (v >= 0.f) atomicMax((int*)addr, __float_as_int(v));
  else          atomicMin((unsigned int*)addr, (unsigned int)__float_as_int(v));
}

__device__ __forceinline__ void edge_nodes(const int* __restrict__ ei, int E, int e,
                                           int& s, int& d) {
  if (e < E) { s = ei[e]; d = ei[E + e]; }
  else       { s = e - E; d = e - E; }      // appended self loops
}

// ---------------- prep ----------------
__global__ void k_split_f16(const float* __restrict__ W, f16* __restrict__ Wh,
                            f16* __restrict__ Wl, int n) {
  int t = blockIdx.x * blockDim.x + threadIdx.x;
  if (t < n) {
    float w = W[t];
    f16 h = (f16)w;
    Wh[t] = h;
    Wl[t] = (f16)(w - (float)h);
  }
}

__global__ void k_init_l1(float* __restrict__ out1, float* __restrict__ m1,
                          float* __restrict__ den1, int N) {
  int t = blockIdx.x * blockDim.x + threadIdx.x;
  if (t < N * F1) out1[t] = 0.f;
  if (t < N * HEADS1) { m1[t] = -INFINITY; den1[t] = 0.f; }
}

__global__ void k_init_l2(float* __restrict__ out2, float* __restrict__ m2,
                          float* __restrict__ den2, int N) {
  int t = blockIdx.x * blockDim.x + threadIdx.x;
  if (t < N * C2_) out2[t] = 0.f;
  if (t < N) { m2[t] = -INFINITY; den2[t] = 0.f; }
}

// ---------------- WMMA GEMM ----------------
// Z[M x Ncols] = (Ah+Al)[M x K] * (Bh+Bl)[K x Ncols], all operands pre-split f16 hi/lo.
// block: 128 rows x NBLK cols, 256 threads = 8 waves, wave w owns rows [w*16, w*16+16).
// precision: ah*bh + ah*bl + al*bh (3 wmma per k-tile per 16x16 subtile) ~ fp32.
// A tiles staged via GLOBAL_LOAD_ASYNC_TO_LDS_B128 when available.
template <int NBLK>
__global__ __launch_bounds__(256) void k_wmma_gemm(
    const f16* __restrict__ Ahg, const f16* __restrict__ Alg,
    const f16* __restrict__ Bhg, const f16* __restrict__ Blg,
    float* __restrict__ Z, int M, int K, int Ncols,
    const float* __restrict__ rank1_vec,   // [M] or null (mod_signal)
    const float* __restrict__ rank1_row)   // [Ncols] or null (last row of W1)
{
  __shared__ f16 Ah[128][LDAH];
  __shared__ f16 Al[128][LDAH];
  __shared__ f16 Bsh[NBLK][LDAH];
  __shared__ f16 Bsl[NBLK][LDAH];

  const int tid  = threadIdx.x;
  const int lane = tid & 31;
  const int wave = tid >> 5;
  const int mb   = blockIdx.x * 128;
  const int nb   = blockIdx.y * NBLK;

  constexpr int NSUB = NBLK / 16;
  v8f acc[NSUB];
#pragma unroll
  for (int i = 0; i < NSUB; ++i)
#pragma unroll
    for (int j = 0; j < 8; ++j) acc[i][j] = 0.f;

  // zero LDS once if this block has tail rows (staging skips rows >= M)
  if (mb + 128 > M) {
    for (int i = tid; i < 128 * LDAH; i += 256) {
      (&Ah[0][0])[i] = (f16)0;
      (&Al[0][0])[i] = (f16)0;
    }
  }

  const int nkt = K / KT;
  for (int kt = 0; kt < nkt; ++kt) {
    __syncthreads();
    // ---- stage A tile 128x32 halves: 4x16B chunks per row, 2 chunks/thread/matrix ----
#pragma unroll
    for (int s = 0; s < 2; ++s) {
      int c   = tid + s * 256;
      int row = c >> 2;
      int seg = c & 3;
      int gr  = mb + row;
      if (gr < M) {
        size_t go = (size_t)gr * K + (size_t)kt * KT + seg * 8;
        copy16_g2l(Ahg + go, &Ah[row][seg * 8]);
        copy16_g2l(Alg + go, &Al[row][seg * 8]);
      }
    }
    // ---- stage B tile 32xNBLK -> LDS transposed [n][k] (small; synchronous) ----
    for (int j = tid; j < KT * NBLK; j += 256) {
      int k = j / NBLK;
      int n = j % NBLK;
      size_t gi = (size_t)(kt * KT + k) * Ncols + nb + n;
      Bsh[n][k] = Bhg[gi];
      Bsl[n][k] = Blg[gi];
    }
    // ---- prefetch next k-tile (global_prefetch_b8) ----
    if (kt + 1 < nkt) {
      int row = tid >> 2, seg = tid & 3;
      int gr = mb + row;
      if (gr < M)
        __builtin_prefetch(Ahg + (size_t)gr * K + (size_t)(kt + 1) * KT + seg * 8, 0, 3);
      __builtin_prefetch(Bhg + (size_t)((kt + 1) * KT + (tid & 31)) * Ncols + nb, 0, 3);
    }
    async_wait_all();
    __syncthreads();

    // ---- A fragments (16x32 f16): lane holds row = lane&15, K chunks per ISA layout ----
    const int arow = wave * 16 + (lane & 15);
    const int kb   = (lane >> 4) * 8;
    v8h a0h = *(const v8h*)&Ah[arow][kb];
    v8h a1h = *(const v8h*)&Ah[arow][kb + 16];
    v8h a0l = *(const v8h*)&Al[arow][kb];
    v8h a1l = *(const v8h*)&Al[arow][kb + 16];
    v16h afh = __builtin_shufflevector(a0h, a1h, 0,1,2,3,4,5,6,7,8,9,10,11,12,13,14,15);
    v16h afl = __builtin_shufflevector(a0l, a1l, 0,1,2,3,4,5,6,7,8,9,10,11,12,13,14,15);

    const int bkb = (lane >> 4) * 8;
#pragma unroll
    for (int t = 0; t < NSUB; ++t) {
      int bn = t * 16 + (lane & 15);
      v8h b0h = *(const v8h*)&Bsh[bn][bkb];
      v8h b1h = *(const v8h*)&Bsh[bn][bkb + 16];
      v8h b0l = *(const v8h*)&Bsl[bn][bkb];
      v8h b1l = *(const v8h*)&Bsl[bn][bkb + 16];
      v16h bfh = __builtin_shufflevector(b0h, b1h, 0,1,2,3,4,5,6,7,8,9,10,11,12,13,14,15);
      v16h bfl = __builtin_shufflevector(b0l, b1l, 0,1,2,3,4,5,6,7,8,9,10,11,12,13,14,15);
      acc[t] = __builtin_amdgcn_wmma_f32_16x16x32_f16(false, afh, false, bfh, (short)0, acc[t], false, false);
      acc[t] = __builtin_amdgcn_wmma_f32_16x16x32_f16(false, afh, false, bfl, (short)0, acc[t], false, false);
      acc[t] = __builtin_amdgcn_wmma_f32_16x16x32_f16(false, afl, false, bfh, (short)0, acc[t], false, false);
    }
  }

  // ---- epilogue: D layout -> (m = r + (lane>>4)*8, n = lane&15); exact fp32 rank-1 for K=257 column ----
#pragma unroll
  for (int t = 0; t < NSUB; ++t) {
    int col = nb + t * 16 + (lane & 15);
#pragma unroll
    for (int r = 0; r < 8; ++r) {
      int m = mb + wave * 16 + (lane >> 4) * 8 + r;
      if (m < M) {
        float v = acc[t][r];
        if (rank1_vec) v += rank1_vec[m] * rank1_row[col];
        Z[(size_t)m * Ncols + col] = v;
      }
    }
  }
}

// ---------------- attention scalars: a_s[n,h] = <z[n,h,:], att_s[h,:]> ----------------
template <int C>
__global__ void k_att_scores(const float* __restrict__ Z, const float* __restrict__ att_s,
                             const float* __restrict__ att_d, float* __restrict__ as_,
                             float* __restrict__ ad_, int N, int H) {
  int t = blockIdx.x * blockDim.x + threadIdx.x;
  if (t >= N * H) return;
  int h = t % H, n = t / H;
  const float* z = Z + (size_t)n * H * C + h * C;
  const float* s = att_s + h * C;
  const float* d = att_d + h * C;
  float accs = 0.f, accd = 0.f;
#pragma unroll 4
  for (int c = 0; c < C; c += 4) {
    float4 zv = *(const float4*)(z + c);
    float4 sv = *(const float4*)(s + c);
    float4 dv = *(const float4*)(d + c);
    accs += zv.x * sv.x + zv.y * sv.y + zv.z * sv.z + zv.w * sv.w;
    accd += zv.x * dv.x + zv.y * dv.y + zv.z * dv.z + zv.w * dv.w;
  }
  as_[t] = accs;
  ad_[t] = accd;
}

// ---------------- edge pass 1: segment max ----------------
__global__ void k_edge_max(const int* __restrict__ ei, int E, int Etot, int H,
                           const float* __restrict__ as_, const float* __restrict__ ad_,
                           float* __restrict__ mbuf) {
  int t = blockIdx.x * blockDim.x + threadIdx.x;
  if (t >= Etot * H) return;
  int h = t % H, e = t / H;
  int s, d; edge_nodes(ei, E, e, s, d);
  float ev = leaky02(as_[s * H + h] + ad_[d * H + h]);
  atomicMaxF(&mbuf[d * H + h], ev);
}

// ---------------- edge pass 2: exp + segment sum (stores exp for reuse) ----------------
__global__ void k_edge_exp(const int* __restrict__ ei, int E, int Etot, int H,
                           const float* __restrict__ as_, const float* __restrict__ ad_,
                           const float* __restrict__ mbuf, float* __restrict__ exws,
                           float* __restrict__ den) {
  int t = blockIdx.x * blockDim.x + threadIdx.x;
  if (t >= Etot * H) return;
  int h = t % H, e = t / H;
  int s, d; edge_nodes(ei, E, e, s, d);
  float ev = leaky02(as_[s * H + h] + ad_[d * H + h]);
  float ex = expf(ev - mbuf[d * H + h]);
  exws[(size_t)e * H + h] = ex;
  atomicAdd(&den[d * H + h], ex);
}

__global__ void k_recip(float* __restrict__ den, int n) {
  int t = blockIdx.x * blockDim.x + threadIdx.x;
  if (t < n) den[t] = 1.f / den[t];
}

// ---------------- edge pass 3: out[dst] += z[src] * alpha (thread = (edge, head, 4 channels)) ----------------
template <int C>
__global__ void k_aggregate(const int* __restrict__ ei, int E, int Etot, int H,
                            const float* __restrict__ exws, const float* __restrict__ invden,
                            const float* __restrict__ Z, float* __restrict__ out) {
  const int C4 = C / 4;
  int t = blockIdx.x * blockDim.x + threadIdx.x;
  if (t >= Etot * H * C4) return;
  int q = t % C4;
  int r = t / C4;
  int h = r % H;
  int e = r / H;
  int s, d; edge_nodes(ei, E, e, s, d);
  float alpha = exws[(size_t)e * H + h] * invden[d * H + h];
  float4 zv = *(const float4*)(Z + (size_t)s * H * C + h * C + q * 4);
  float* o = out + (size_t)d * H * C + h * C + q * 4;
  atomicAdd(o + 0, zv.x * alpha);
  atomicAdd(o + 1, zv.y * alpha);
  atomicAdd(o + 2, zv.z * alpha);
  atomicAdd(o + 3, zv.w * alpha);
}

// ---------------- epilogues ----------------
// BN + ELU, emitting h1 pre-split to f16 hi/lo (feeds GEMM2's async staging)
__global__ void k_post1(const float* __restrict__ out1, const float* __restrict__ bias,
                        const float* __restrict__ g, const float* __restrict__ b,
                        const float* __restrict__ mu, const float* __restrict__ var,
                        f16* __restrict__ h1h, f16* __restrict__ h1l, int N) {
  int t = blockIdx.x * blockDim.x + threadIdx.x;
  if (t >= N * F1) return;
  int c = t % F1;
  float v = out1[t] + bias[c];
  v = (v - mu[c]) * (g[c] * rsqrtf(var[c] + 1e-5f)) + b[c];
  v = v > 0.f ? v : (expf(v) - 1.f);      // ELU
  f16 h = (f16)v;
  h1h[t] = h;
  h1l[t] = (f16)(v - (float)h);
}

__global__ void k_post2(const float* __restrict__ out2, const float* __restrict__ bias,
                        const float* __restrict__ g, const float* __restrict__ b,
                        const float* __restrict__ mu, const float* __restrict__ var,
                        float* __restrict__ emb, int N) {
  int t = blockIdx.x * blockDim.x + threadIdx.x;
  if (t >= N * C2_) return;
  int c = t % C2_;
  float v = out2[t] + bias[c];
  emb[t] = (v - mu[c]) * (g[c] * rsqrtf(var[c] + 1e-5f)) + b[c];
}

// ---------------- scorer: sigmoid(relu([emb[src]|emb[n]] @ Ws1 + bs1) @ Ws2 + bs2) ----------------
__global__ void k_score(const float* __restrict__ emb, const int* __restrict__ midx_p,
                        const float* __restrict__ Ws1, const float* __restrict__ bs1,
                        const float* __restrict__ Ws2, const float* __restrict__ bs2,
                        float* __restrict__ out, int N) {
  __shared__ float sW[64 * 32];
  __shared__ float sSrc[32];
  __shared__ float sBase[32];
  __shared__ float sW2[32];
  int tid = threadIdx.x;
  int midx = midx_p[0];
  for (int i = tid; i < 64 * 32; i += blockDim.x) sW[i] = Ws1[i];
  if (tid < 32) {
    sSrc[tid] = emb[(size_t)midx * 32 + tid];
    sW2[tid]  = Ws2[tid];
  }
  __syncthreads();
  if (tid < 32) {
    float a = bs1[tid];
#pragma unroll
    for (int k = 0; k < 32; ++k) a += sSrc[k] * sW[k * 32 + tid];
    sBase[tid] = a;
  }
  __syncthreads();
  int n = blockIdx.x * blockDim.x + tid;
  if (n >= N) return;
  float en[32];
#pragma unroll
  for (int k = 0; k < 32; ++k) en[k] = emb[(size_t)n * 32 + k];
  float s = bs2[0];
#pragma unroll
  for (int j = 0; j < 32; ++j) {
    float a = sBase[j];
#pragma unroll
    for (int k = 0; k < 32; ++k) a += en[k] * sW[(32 + k) * 32 + j];
    a = a > 0.f ? a : 0.f;
    s += a * sW2[j];
  }
  out[n] = 1.f / (1.f + expf(-s));
}

// ---------------- host ----------------
extern "C" void kernel_launch(void* const* d_in, const int* in_sizes, int n_in,
                              void* d_out, int out_size, void* d_ws, size_t ws_size,
                              hipStream_t stream) {
  const float* x      = (const float*)d_in[0];
  const int*   ei     = (const int*)d_in[1];
  const float* mod    = (const float*)d_in[2];
  const int*   midx   = (const int*)d_in[3];
  const float* W1     = (const float*)d_in[6];
  const float* att_s1 = (const float*)d_in[7];
  const float* att_d1 = (const float*)d_in[8];
  const float* bias1  = (const float*)d_in[9];
  const float* g1 = (const float*)d_in[10], *b1 = (const float*)d_in[11];
  const float* mu1 = (const float*)d_in[12], *v1 = (const float*)d_in[13];
  const float* W2     = (const float*)d_in[14];
  const float* att_s2 = (const float*)d_in[15];
  const float* att_d2 = (const float*)d_in[16];
  const float* bias2  = (const float*)d_in[17];
  const float* g2 = (const float*)d_in[18], *b2 = (const float*)d_in[19];
  const float* mu2 = (const float*)d_in[20], *v2 = (const float*)d_in[21];
  const float* Ws1 = (const float*)d_in[22], *bs1 = (const float*)d_in[23];
  const float* Ws2 = (const float*)d_in[24], *bs2 = (const float*)d_in[25];

  const int N    = in_sizes[2];        // mod_signal length
  const int E    = in_sizes[1] / 2;
  const int Etot = E + N;

  // ----- workspace arena (aliased; ~96 MB total) -----
  char* p = (char*)d_ws;
  auto alloc = [&](size_t bytes) -> char* {
    char* r = p;
    p += (bytes + 255) & ~(size_t)255;
    return r;
  };
  f16* W1h = (f16*)alloc((size_t)DIN * F1 * sizeof(f16));
  f16* W1l = (f16*)alloc((size_t)DIN * F1 * sizeof(f16));
  f16* W2h = (f16*)alloc((size_t)F1 * C2_ * sizeof(f16));
  f16* W2l = (f16*)alloc((size_t)F1 * C2_ * sizeof(f16));
  float* as1  = (float*)alloc((size_t)N * HEADS1 * 4);
  float* ad1  = (float*)alloc((size_t)N * HEADS1 * 4);
  float* m1b  = (float*)alloc((size_t)N * HEADS1 * 4);
  float* den1 = (float*)alloc((size_t)N * HEADS1 * 4);
  float* z1   = (float*)alloc((size_t)N * F1 * 4);   // region A: z1, later h1h+h1l (f16)
  float* out1 = (float*)alloc((size_t)N * F1 * 4);   // region B: xh+xl (f16) early, out1 after GEMM1
  size_t ex1_bytes = (size_t)Etot * HEADS1 * 4;
  size_t l2_bytes  = (size_t)N * C2_ * 4 * 3 + (size_t)Etot * 4 + (size_t)N * 4 * 4;
  char* regC = alloc(ex1_bytes > l2_bytes ? ex1_bytes : l2_bytes);
  float* ex1 = (float*)regC;                         // dead after layer-1 aggregate
  // layer-2 overlay of regC (written only after ex1's last read):
  float* z2   = (float*)regC;
  float* out2 = z2 + (size_t)N * C2_;
  float* emb  = out2 + (size_t)N * C2_;
  float* ex2  = emb + (size_t)N * C2_;
  float* as2  = ex2 + Etot;
  float* ad2  = as2 + N;
  float* m2b  = ad2 + N;
  float* den2 = m2b + N;
  // region-B overlay: x pre-split hi/lo (dead once GEMM1 completes; out1 zeroed after)
  f16* xh = (f16*)out1;
  f16* xl = xh + (size_t)N * DIN;
  // region-A overlay: h1 pre-split hi/lo (written by k_post1 after z1 is dead)
  f16* h1h = (f16*)z1;
  f16* h1l = h1h + (size_t)N * F1;

  const int B = 256;
  auto cdiv = [](long long a, long long b) { return (int)((a + b - 1) / b); };

  // ----- prep: split operands to f16 hi/lo -----
  k_split_f16<<<cdiv((size_t)DIN * F1, B), B, 0, stream>>>(W1, W1h, W1l, DIN * F1);
  k_split_f16<<<cdiv((size_t)F1 * C2_, B), B, 0, stream>>>(W2, W2h, W2l, F1 * C2_);
  k_split_f16<<<cdiv((size_t)N * DIN, B), B, 0, stream>>>(x, xh, xl, N * DIN);

  // ----- layer 1 -----
  dim3 gg1(cdiv(N, 128), F1 / 64);
  k_wmma_gemm<64><<<gg1, B, 0, stream>>>(xh, xl, W1h, W1l, z1, N, DIN, F1,
                                         mod, W1 + (size_t)DIN * F1);
  // xh/xl dead; now safe to zero out1 (same region) + layer-1 accumulators
  k_init_l1<<<cdiv((size_t)N * F1, B), B, 0, stream>>>(out1, m1b, den1, N);
  k_att_scores<C1_><<<cdiv((size_t)N * HEADS1, B), B, 0, stream>>>(z1, att_s1, att_d1,
                                                                   as1, ad1, N, HEADS1);
  k_edge_max<<<cdiv((size_t)Etot * HEADS1, B), B, 0, stream>>>(ei, E, Etot, HEADS1, as1, ad1, m1b);
  k_edge_exp<<<cdiv((size_t)Etot * HEADS1, B), B, 0, stream>>>(ei, E, Etot, HEADS1, as1, ad1,
                                                               m1b, ex1, den1);
  k_recip<<<cdiv((size_t)N * HEADS1, B), B, 0, stream>>>(den1, N * HEADS1);
  k_aggregate<C1_><<<cdiv((long long)Etot * HEADS1 * (C1_ / 4), B), B, 0, stream>>>(
      ei, E, Etot, HEADS1, ex1, den1, z1, out1);
  k_post1<<<cdiv((size_t)N * F1, B), B, 0, stream>>>(out1, bias1, g1, b1, mu1, v1,
                                                     h1h, h1l, N);

  // ----- layer 2 (accumulators live in regC; init AFTER ex1's last use) -----
  k_init_l2<<<cdiv((size_t)N * C2_, B), B, 0, stream>>>(out2, m2b, den2, N);
  dim3 gg2(cdiv(N, 128), 1);
  k_wmma_gemm<32><<<gg2, B, 0, stream>>>(h1h, h1l, W2h, W2l, z2, N, F1, C2_,
                                         nullptr, nullptr);
  k_att_scores<C2_><<<cdiv(N, B), B, 0, stream>>>(z2, att_s2, att_d2, as2, ad2, N, 1);
  k_edge_max<<<cdiv(Etot, B), B, 0, stream>>>(ei, E, Etot, 1, as2, ad2, m2b);
  k_edge_exp<<<cdiv(Etot, B), B, 0, stream>>>(ei, E, Etot, 1, as2, ad2, m2b, ex2, den2);
  k_recip<<<cdiv(N, B), B, 0, stream>>>(den2, N);
  k_aggregate<C2_><<<cdiv((long long)Etot * (C2_ / 4), B), B, 0, stream>>>(
      ei, E, Etot, 1, ex2, den2, z2, out2);
  k_post2<<<cdiv((size_t)N * C2_, B), B, 0, stream>>>(out2, bias2, g2, b2, mu2, v2, emb, N);

  // ----- scorer -----
  k_score<<<cdiv(N, B), B, 0, stream>>>(emb, midx, Ws1, bs1, Ws2, bs2, (float*)d_out, N);
}
